// ReformerBlockPreLN_51479478010642
// MI455X (gfx1250) — matrix operand order repaired and checked by reading, not compiled, not run on hardware
//
#include <hip/hip_runtime.h>
#include <hip/hip_bf16.h>
#include <math.h>

typedef _Float16 half_t;
typedef __attribute__((ext_vector_type(16))) _Float16      v16h;
typedef __attribute__((ext_vector_type(8)))  float         v8f;
typedef __attribute__((ext_vector_type(4)))  unsigned int  u32x4;
typedef __attribute__((ext_vector_type(8)))  int           i32x8;
typedef __attribute__((ext_vector_type(4)))  int           i32x4;

#define B_     2
#define S_     4096
#define D_     1024
#define H_     16
#define DH_    64
#define CHUNK_ 128
#define NC_    (S_ / CHUNK_)        // 32 chunks
#define RNDS_  2
#define NBUCK_ 64
#define MLP_   4096
#define NROW_  (B_ * H_ * RNDS_)    // 64 independent attention rows
#define BS_    (B_ * S_)            // 8192 tokens

#define BT_LD   40                  // BsT leading dim (halves): 80B rows, 16B aligned
#define VT_LD   392                 // vT leading dim (halves): 784B rows, 16B aligned
#define KVBUF_  (VT_LD * DH_)       // 25088 halves >= 384*64, reused K then V^T

#if __has_builtin(__builtin_amdgcn_tensor_load_to_lds)
#define USE_TDM 1
#else
#define USE_TDM 0
#endif

// ---------------------------------------------------------------- WMMA helper
__device__ inline v8f wmma_f16(v16h a, v16h b, v8f c) {
  return __builtin_amdgcn_wmma_f32_16x16x32_f16(false, a, false, b, (short)0, c,
                                                false, false);
}

union FragU { u32x4 q[2]; v16h v; };
union H8   { u32x4 q; half_t h[8]; };
union Pk2  { unsigned int w; half_t h[2]; };

// 16 contiguous halves -> fragment (two b128 loads)
__device__ inline v16h load_frag16(const half_t* p) {
  FragU u;
  u.q[0] = *(const u32x4*)(p);
  u.q[1] = *(const u32x4*)(p + 8);
  return u.v;
}

// A-fragment (16x32, row-major, ld halves): halves 0..15 at row*ld + hi*16
// (contiguous K order, matching the B fragments)
__device__ inline v16h load_afrag_rm(const half_t* src, int ld) {
  int lane = threadIdx.x & 31;
  int row = lane & 15, hi = (lane >> 4) & 1;
  FragU u;
  u.q[0] = *(const u32x4*)(src + row * ld + hi * 16);
  u.q[1] = *(const u32x4*)(src + row * ld + hi * 16 + 8);
  return u.v;
}

// B-fragment from K-contiguous (transposed) storage: lane col, K = hi*16..+15
__device__ inline v16h load_bfrag_kc(const half_t* srcT, int ldT) {
  int lane = threadIdx.x & 31;
  int col = lane & 15, hi = (lane >> 4) & 1;
  return load_frag16(srcT + col * ldT + hi * 16);
}

#if USE_TDM
// Tensor Data Mover: DMA a 2-D f16 tile [tile_rows x tile_k] from a row-major
// [tensor_rows x tensor_k] tensor into contiguous LDS. D# groups per ISA ch.8.
__device__ inline void tdm_load_2d_f16(unsigned int lds_off, const void* gaddr,
                                       unsigned int tile_k, unsigned int tile_rows,
                                       unsigned int tensor_k, unsigned int tensor_rows) {
  unsigned long long ga = (unsigned long long)(size_t)gaddr;
  u32x4 g0;
  g0[0] = 1u;                                     // count=1 (valid user D#)
  g0[1] = lds_off;                                // lds_addr (bytes)
  g0[2] = (unsigned int)(ga & 0xFFFFFFFFu);       // global_addr[31:0]
  g0[3] = (unsigned int)((ga >> 32) & 0x1FFFFFFu) // global_addr[56:32]
          | (2u << 30);                           // type=2 ("image")
  i32x8 g1;
  g1[0] = (int)(1u << 16);                        // data_size=1 -> 2 bytes/elem
  g1[1] = (int)((tensor_k & 0xFFFFu) << 16);      // tensor_dim0[15:0]
  g1[2] = (int)(((tensor_k >> 16) & 0xFFFFu) |
                ((tensor_rows & 0xFFFFu) << 16)); // dim0[31:16] | dim1[15:0]
  g1[3] = (int)(((tensor_rows >> 16) & 0xFFFFu) |
                ((tile_k & 0xFFFFu) << 16));      // dim1[31:16] | tile_dim0
  g1[4] = (int)(tile_rows & 0xFFFFu);             // tile_dim1 (tile_dim2=0)
  g1[5] = (int)tensor_k;                          // tensor_dim0_stride[31:0]
  g1[6] = 0;                                      // stride0[47:32], stride1[15:0]
  g1[7] = 0;
  i32x4 z4 = {0, 0, 0, 0};
#if defined(__clang_major__) && __clang_major__ >= 23
  i32x8 z8 = {0, 0, 0, 0, 0, 0, 0, 0};
  __builtin_amdgcn_tensor_load_to_lds(g0, g1, z4, z4, z8, 0);
#else
  __builtin_amdgcn_tensor_load_to_lds(g0, g1, z4, z4, 0);
#endif
}

__device__ inline void tdm_wait() {
#if __has_builtin(__builtin_amdgcn_s_wait_tensorcnt)
  __builtin_amdgcn_s_wait_tensorcnt(0);
#else
  asm volatile("s_wait_tensorcnt 0x0" ::: "memory");
#endif
}
#endif  // USE_TDM

// ---------------------------------------------------------------- small utils
__global__ void f32_to_f16_kernel(const float* __restrict__ src,
                                  half_t* __restrict__ dst, int n) {
  int i = blockIdx.x * blockDim.x + threadIdx.x;
  if (i < n) dst[i] = (half_t)src[i];
}

// LayerNorm over D=1024, one block (256 threads) per token. f32 in, f16 out.
__global__ __launch_bounds__(256)
void ln_kernel(const float* __restrict__ x, const float* __restrict__ scale,
               const float* __restrict__ bias, half_t* __restrict__ y) {
  __shared__ float red[256];
  int tok = blockIdx.x;
  int tid = threadIdx.x;
  const float* xp = x + (size_t)tok * D_;
  float vals[4];
  float s = 0.f;
#pragma unroll
  for (int k = 0; k < 4; ++k) { vals[k] = xp[tid + k * 256]; s += vals[k]; }
  red[tid] = s;
  __syncthreads();
  for (int o = 128; o > 0; o >>= 1) {
    if (tid < o) red[tid] += red[tid + o];
    __syncthreads();
  }
  float mu = red[0] * (1.f / D_);
  __syncthreads();
  float vsum = 0.f;
#pragma unroll
  for (int k = 0; k < 4; ++k) { float d = vals[k] - mu; vsum += d * d; }
  red[tid] = vsum;
  __syncthreads();
  for (int o = 128; o > 0; o >>= 1) {
    if (tid < o) red[tid] += red[tid + o];
    __syncthreads();
  }
  float inv = rsqrtf(red[0] * (1.f / D_) + 1e-6f);
#pragma unroll
  for (int k = 0; k < 4; ++k) {
    int d = tid + k * 256;
    y[(size_t)tok * D_ + d] = (half_t)((vals[k] - mu) * inv * scale[d] + bias[d]);
  }
}

// ---------------------------------------------------------------- WMMA GEMM
// C[M,N] = A[M,K] * B[K,N]; A,B f16 row-major. 128x128 block tile, 8 waves,
// each wave owns a 32x64 sub-tile (2x4 v_wmma_f32_16x16x32_f16 accumulators).
// A slab is DMA'd by the Tensor Data Mover (TENSORcnt); B slab is staged
// transposed so B-fragments are two ds_load_b128.
enum { MODE_F16 = 0, MODE_RES_F32 = 1, MODE_BIAS_GELU_F16 = 2, MODE_BIAS_RES_F32 = 3 };

__global__ __launch_bounds__(256)
void gemm_wmma_kernel(const half_t* __restrict__ A, const half_t* __restrict__ Bm,
                      const float* __restrict__ bias, const float* __restrict__ resid,
                      float* __restrict__ Cf, half_t* __restrict__ Ch,
                      int M, int N, int K, int mode) {
  __shared__ __align__(16) half_t As[128 * 32];      // [m][k] row-major
  __shared__ __align__(16) half_t BsT[128 * BT_LD];  // [n][k] transposed
  int tid = threadIdx.x, lane = tid & 31, wave = tid >> 5;
  int colc = lane & 15, hi = (lane >> 4) & 1;
  int wm = wave & 3, wn = wave >> 2;                 // 4 (M) x 2 (N) wave grid
  int bm = blockIdx.y * 128, bn = blockIdx.x * 128;

  v8f acc[2][4];
#pragma unroll
  for (int i = 0; i < 2; ++i)
#pragma unroll
    for (int j = 0; j < 4; ++j) {
      v8f z = {0.f, 0.f, 0.f, 0.f, 0.f, 0.f, 0.f, 0.f};
      acc[i][j] = z;
    }

#if USE_TDM
  unsigned int as_lds = (unsigned int)(size_t)(void*)&As[0];
#endif

  for (int k0 = 0; k0 < K; k0 += 32) {
#if USE_TDM
    // one wave drives the TDM: A tile 128 rows x 32 halves, row pitch K
    if (wave == 0) {
      tdm_load_2d_f16(as_lds, &A[(size_t)bm * K + k0], 32u, 128u,
                      (unsigned int)K, (unsigned int)M);
    }
#else
#pragma unroll
    for (int u0 = 0; u0 < 2; ++u0) {
      int u = tid + u0 * 256;
      int r = u >> 2, c8 = (u & 3) * 8;
      *(u32x4*)&As[r * 32 + c8] =
          *(const u32x4*)&A[(size_t)(bm + r) * K + k0 + c8];
    }
#endif
    // stage B 32x128 transposed: each thread owns a 2(k) x 8(n) subtile,
    // two b128 global loads, packed b32 LDS stores
    {
      int kk2 = (tid >> 4) * 2;
      int n8  = (tid & 15) * 8;
      H8 a0, a1;
      a0.q = *(const u32x4*)&Bm[(size_t)(k0 + kk2) * N + bn + n8];
      a1.q = *(const u32x4*)&Bm[(size_t)(k0 + kk2 + 1) * N + bn + n8];
#pragma unroll
      for (int j = 0; j < 8; ++j) {
        Pk2 pk;
        pk.h[0] = a0.h[j];
        pk.h[1] = a1.h[j];
        *(unsigned int*)&BsT[(n8 + j) * BT_LD + kk2] = pk.w;
      }
    }
    if (k0 + 32 < K) {  // hint next B slab into cache (global_prefetch_b8)
      __builtin_prefetch(&Bm[(size_t)(k0 + 32 + (tid >> 3)) * N + bn], 0, 0);
    }
#if USE_TDM
    if (wave == 0) tdm_wait();
#endif
    __syncthreads();

    v16h af[2], bf[4];
#pragma unroll
    for (int i = 0; i < 2; ++i)
      af[i] = load_afrag_rm(&As[(wm * 32 + i * 16) * 32], 32);
#pragma unroll
    for (int j = 0; j < 4; ++j)
      bf[j] = load_bfrag_kc(&BsT[(wn * 64 + j * 16) * BT_LD], BT_LD);
#pragma unroll
    for (int i = 0; i < 2; ++i)
#pragma unroll
      for (int j = 0; j < 4; ++j) acc[i][j] = wmma_f16(af[i], bf[j], acc[i][j]);
    __syncthreads();
  }

  int col0 = bn + wn * 64;
#pragma unroll
  for (int i = 0; i < 2; ++i)
#pragma unroll
    for (int j = 0; j < 4; ++j)
#pragma unroll
      for (int r = 0; r < 8; ++r) {
        int rowi = bm + wm * 32 + i * 16 + hi * 8 + r;
        int col  = col0 + j * 16 + colc;
        size_t o = (size_t)rowi * N + col;
        float a = acc[i][j][r];
        if (mode == MODE_F16) {
          Ch[o] = (half_t)a;
        } else if (mode == MODE_RES_F32) {
          Cf[o] = a + resid[o];
        } else if (mode == MODE_BIAS_GELU_F16) {
          float xx = a + bias[col];
          float g = 0.5f * xx *
                    (1.f + tanhf(0.7978845608f * (xx + 0.044715f * xx * xx * xx)));
          Ch[o] = (half_t)g;
        } else {  // MODE_BIAS_RES_F32
          Cf[o] = a + bias[col] + resid[o];
        }
      }
}

// ---------------------------------------------------------------- LSH buckets
__global__ __launch_bounds__(256)
void bucket_kernel(const half_t* __restrict__ qk, const float* __restrict__ rot,
                   half_t* __restrict__ xn, unsigned short* __restrict__ buckets) {
  int idx = blockIdx.x * blockDim.x + threadIdx.x;  // B*H*S
  int s = idx & (S_ - 1);
  int t = idx >> 12;
  int h = t & (H_ - 1);
  int b = t >> 4;
  const half_t* q = qk + (((size_t)(b * S_ + s)) * H_ + h) * DH_;
  float qf[DH_];
  float ss = 0.f;
#pragma unroll
  for (int d = 0; d < DH_; ++d) { qf[d] = (float)q[d]; ss += qf[d] * qf[d]; }
  float inv = 1.f / (sqrtf(ss) + 1e-6f);
  half_t* xp = xn + (((size_t)(b * S_ + s)) * H_ + h) * DH_;
#pragma unroll
  for (int d = 0; d < DH_; ++d) { qf[d] *= inv; xp[d] = (half_t)qf[d]; }
#pragma unroll
  for (int r = 0; r < RNDS_; ++r) {
    float best = -1e30f;
    int bi = 0;
    for (int m = 0; m < NBUCK_ / 2; ++m) {
      float dot = 0.f;
#pragma unroll
      for (int d = 0; d < DH_; ++d)
        dot += qf[d] * rot[d * (RNDS_ * (NBUCK_ / 2)) + r * (NBUCK_ / 2) + m];
      if (dot > best)  { best = dot;  bi = m; }
      if (-dot > best) { best = -dot; bi = m + NBUCK_ / 2; }
    }
    buckets[((size_t)((b * H_ + h) * RNDS_ + r)) * S_ + s] = (unsigned short)bi;
  }
}

// ---------------------------------------------------------------- stable sort
// Per row: stable counting sort of 4096 six-bit keys (256 segments of 16).
__global__ __launch_bounds__(256)
void sort_kernel(const unsigned short* __restrict__ buckets,
                 unsigned int* __restrict__ order) {
  __shared__ unsigned short hist[256][NBUCK_];  // 32KB
  __shared__ unsigned int totals[NBUCK_];
  __shared__ unsigned int bbase[NBUCK_];
  int row = blockIdx.x;
  int tid = threadIdx.x;
  const unsigned short* bk = buckets + (size_t)row * S_;
  unsigned int* ord = order + (size_t)row * S_;
#pragma unroll
  for (int b = 0; b < NBUCK_; ++b) hist[tid][b] = 0;
  __syncthreads();
  int base = tid * 16;
  unsigned short local[16];
#pragma unroll
  for (int i = 0; i < 16; ++i) {
    local[i] = bk[base + i];
    hist[tid][local[i]]++;
  }
  __syncthreads();
  if (tid < NBUCK_) {
    unsigned int run = 0;
    for (int sgm = 0; sgm < 256; ++sgm) {
      unsigned short v = hist[sgm][tid];
      hist[sgm][tid] = (unsigned short)run;
      run += v;
    }
    totals[tid] = run;
  }
  __syncthreads();
  if (tid == 0) {
    unsigned int a = 0;
    for (int b = 0; b < NBUCK_; ++b) { bbase[b] = a; a += totals[b]; }
  }
  __syncthreads();
#pragma unroll
  for (int i = 0; i < 16; ++i) {
    int key = local[i];
    unsigned int dest = bbase[key] + hist[tid][key];
    hist[tid][key]++;
    ord[dest] = (unsigned int)(base + i);
  }
}

// ---------------------------------------------------------------- attention
// One block (8 waves) per (row, chunk). Each wave owns 16 query rows.
// dots = Q(16x64) x K^T via WMMA (K buffer is K-contiguous per key);
// softmax with half-wave shuffles; V re-staged transposed into the same LDS
// buffer so P.V B-fragments are two ds_load_b128; P relayout goes through a
// per-wave LDS scratch guarded by s_wait_dscnt.
__global__ __launch_bounds__(256)
void attn_kernel(const half_t* __restrict__ qk, const half_t* __restrict__ xn,
                 const half_t* __restrict__ v, const unsigned int* __restrict__ order,
                 half_t* __restrict__ o_out, float* __restrict__ slog_out) {
  __shared__ __align__(16) half_t kvbuf[KVBUF_];     // ~49KB: K rows, then V^T
  __shared__ int bt[3 * CHUNK_];
  __shared__ __align__(16) half_t ps[8][16 * 32];    // 8KB P scratch
  int row = blockIdx.y;
  int c   = blockIdx.x;
  int b = row >> 5;                // row / (H*RNDS)
  int h = (row >> 1) & (H_ - 1);
  const unsigned int* ord = order + (size_t)row * S_;
  int tid = threadIdx.x, lane = tid & 31, wave = tid >> 5;
  int colc = lane & 15, hi = (lane >> 4) & 1;
  int cp = (c + NC_ - 1) & (NC_ - 1);
  int cn = (c + 1) & (NC_ - 1);

  // stage K (length-normalized shared-QK) for [own, prev, next]: row-major [kk][d]
  for (int j = tid; j < 3 * CHUNK_; j += 256) {
    int cc = (j < CHUNK_) ? c : ((j < 2 * CHUNK_) ? cp : cn);
    int p  = cc * CHUNK_ + (j & (CHUNK_ - 1));
    int tok = (int)ord[p];
    bt[j] = tok;
    const u32x4* src = (const u32x4*)(xn + (((size_t)(b * S_ + tok)) * H_ + h) * DH_);
    u32x4* dst = (u32x4*)(kvbuf + (size_t)j * DH_);
#pragma unroll
    for (int t = 0; t < 8; ++t) dst[t] = src[t];
  }
  __syncthreads();

  // Q fragments (A layout): contiguous b128 pairs gathered via the permutation
  int qrow = wave * 16 + colc;
  int qtokA = bt[qrow];
  const half_t* qp = qk + (((size_t)(b * S_ + qtokA)) * H_ + h) * DH_;
  FragU uq0, uq1;
  uq0.q[0] = *(const u32x4*)(qp + hi * 16);
  uq0.q[1] = *(const u32x4*)(qp + hi * 16 + 8);
  uq1.q[0] = *(const u32x4*)(qp + 32 + hi * 16);
  uq1.q[1] = *(const u32x4*)(qp + 40 + hi * 16);
  v16h aq0 = uq0.v, aq1 = uq1.v;

  // dots: 24 N-tiles of 16 keys, two K=32 WMMAs each
  v8f dacc[24];
#pragma unroll
  for (int nt = 0; nt < 24; ++nt) {
    v8f z = {0.f, 0.f, 0.f, 0.f, 0.f, 0.f, 0.f, 0.f};
    dacc[nt] = z;
  }
#pragma unroll
  for (int nt = 0; nt < 24; ++nt) {
    const half_t* kp = kvbuf + (size_t)(nt * 16 + colc) * DH_;
    v16h b0 = load_frag16(kp + hi * 16);
    v16h b1 = load_frag16(kp + 32 + hi * 16);
    dacc[nt] = wmma_f16(aq0, b0, dacc[nt]);
    dacc[nt] = wmma_f16(aq1, b1, dacc[nt]);
  }
  __syncthreads();

  // re-stage V transposed into kvbuf: vT[d][kk], ld=VT_LD
  for (int su = tid; su < (3 * CHUNK_ / 2) * (DH_ / 8); su += 256) {  // 1536
    int d8  = (su & 7) * 8;
    int kp2 = (su >> 3) * 2;
    int t0 = bt[kp2], t1 = bt[kp2 + 1];
    H8 a0, a1;
    a0.q = *(const u32x4*)(v + (((size_t)(b * S_ + t0)) * H_ + h) * DH_ + d8);
    a1.q = *(const u32x4*)(v + (((size_t)(b * S_ + t1)) * H_ + h) * DH_ + d8);
#pragma unroll
    for (int j = 0; j < 8; ++j) {
      Pk2 pk;
      pk.h[0] = a0.h[j];
      pk.h[1] = a1.h[j];
      *(unsigned int*)&kvbuf[(d8 + j) * VT_LD + kp2] = pk.w;
    }
  }
  __syncthreads();

  // scale + self-mask + softmax (rows live across 16 lanes of a half-wave)
  int qtk[8];
#pragma unroll
  for (int r = 0; r < 8; ++r) qtk[r] = bt[wave * 16 + hi * 8 + r];
  float mrow[8], srow[8], pinv[8];
#pragma unroll
  for (int r = 0; r < 8; ++r) mrow[r] = -1e30f;
  const float scale = 0.125f;  // 1/sqrt(DH)
#pragma unroll
  for (int nt = 0; nt < 24; ++nt) {
    int ct = bt[nt * 16 + colc];
#pragma unroll
    for (int r = 0; r < 8; ++r) {
      float dv = dacc[nt][r] * scale - ((qtk[r] == ct) ? 1e5f : 0.f);
      dacc[nt][r] = dv;
      mrow[r] = fmaxf(mrow[r], dv);
    }
  }
#pragma unroll
  for (int r = 0; r < 8; ++r)
#pragma unroll
    for (int off = 1; off < 16; off <<= 1)
      mrow[r] = fmaxf(mrow[r], __shfl_xor(mrow[r], off, 32));
#pragma unroll
  for (int r = 0; r < 8; ++r) srow[r] = 0.f;
#pragma unroll
  for (int nt = 0; nt < 24; ++nt)
#pragma unroll
    for (int r = 0; r < 8; ++r) {
      float e = __expf(dacc[nt][r] - mrow[r]);
      dacc[nt][r] = e;
      srow[r] += e;
    }
#pragma unroll
  for (int r = 0; r < 8; ++r) {
#pragma unroll
    for (int off = 1; off < 16; off <<= 1) srow[r] += __shfl_xor(srow[r], off, 32);
    pinv[r] = 1.f / srow[r];
  }
  if (colc == 0) {
#pragma unroll
    for (int r = 0; r < 8; ++r)
      slog_out[(size_t)row * S_ + qtk[r]] = mrow[r] + __logf(srow[r]);
  }

  // O = P x V : 12 K-tiles of 32; P relayout through per-wave LDS scratch
  v8f oacc[4];
#pragma unroll
  for (int j2 = 0; j2 < 4; ++j2) {
    v8f z = {0.f, 0.f, 0.f, 0.f, 0.f, 0.f, 0.f, 0.f};
    oacc[j2] = z;
  }
  half_t* myps = &ps[wave][0];
#pragma unroll
  for (int t = 0; t < 12; ++t) {
#pragma unroll
    for (int hf = 0; hf < 2; ++hf) {
      int nt = 2 * t + hf;
#pragma unroll
      for (int r = 0; r < 8; ++r) {
        int m = hi * 8 + r;
        myps[m * 32 + hf * 16 + colc] = (half_t)(dacc[nt][r] * pinv[r]);
      }
    }
    asm volatile("s_wait_dscnt 0x0" ::: "memory");  // LDS stores -> A-frag loads
    v16h ap = load_afrag_rm(myps, 32);
#pragma unroll
    for (int j2 = 0; j2 < 4; ++j2) {
      const half_t* vp = kvbuf + (size_t)(j2 * 16 + colc) * VT_LD + t * 32 + hi * 16;
      v16h bvf = load_frag16(vp);
      oacc[j2] = wmma_f16(ap, bvf, oacc[j2]);
    }
  }

  // scatter O back to original token order (undo permutation)
#pragma unroll
  for (int j2 = 0; j2 < 4; ++j2)
#pragma unroll
    for (int r = 0; r < 8; ++r)
      o_out[((size_t)row * S_ + qtk[r]) * DH_ + j2 * 16 + colc] = (half_t)oacc[j2][r];
}

// ------------------------------------------------------- combine hash rounds
__global__ __launch_bounds__(256)
void combine_kernel(const half_t* __restrict__ o, const float* __restrict__ slog,
                    half_t* __restrict__ att) {
  int idx = blockIdx.x * blockDim.x + threadIdx.x;  // B*H*S
  int s = idx & (S_ - 1);
  int t = idx >> 12;
  int h = t & (H_ - 1);
  int b = t >> 4;
  int r0 = (b * H_ + h) * RNDS_;
  float l0 = slog[(size_t)r0 * S_ + s];
  float l1 = slog[(size_t)(r0 + 1) * S_ + s];
  float m = fmaxf(l0, l1);
  float w0 = __expf(l0 - m), w1 = __expf(l1 - m);
  float inv = 1.f / (w0 + w1);
  w0 *= inv; w1 *= inv;
  const half_t* o0 = o + ((size_t)r0 * S_ + s) * DH_;
  const half_t* o1 = o + ((size_t)(r0 + 1) * S_ + s) * DH_;
  half_t* dst = att + (((size_t)(b * S_ + s)) * H_ + h) * DH_;
#pragma unroll
  for (int d = 0; d < DH_; ++d)
    dst[d] = (half_t)(w0 * (float)o0[d] + w1 * (float)o1[d]);
}

// ---------------------------------------------------------------- host launch
extern "C" void kernel_launch(void* const* d_in, const int* in_sizes, int n_in,
                              void* d_out, int out_size, void* d_ws, size_t ws_size,
                              hipStream_t stream) {
  (void)in_sizes; (void)n_in; (void)out_size; (void)ws_size;
  const float* inputs    = (const float*)d_in[0];
  const float* ln1_scale = (const float*)d_in[1];
  const float* ln1_bias  = (const float*)d_in[2];
  const float* w_qk      = (const float*)d_in[3];
  const float* w_v       = (const float*)d_in[4];
  const float* w_out     = (const float*)d_in[5];
  const float* rotations = (const float*)d_in[6];
  const float* ln2_scale = (const float*)d_in[7];
  const float* ln2_bias  = (const float*)d_in[8];
  const float* w_mlp1    = (const float*)d_in[9];
  const float* b_mlp1    = (const float*)d_in[10];
  const float* w_mlp2    = (const float*)d_in[11];
  const float* b_mlp2    = (const float*)d_in[12];
  float* out = (float*)d_out;

  char* wsb = (char*)d_ws;
  size_t off = 0;
  auto alloc = [&](size_t bytes) -> char* {
    char* p = wsb + off;
    off = (off + bytes + 255) & ~(size_t)255;
    return p;
  };

  half_t*        xln    = (half_t*)alloc((size_t)BS_ * D_ * 2);
  half_t*        wqk_h  = (half_t*)alloc((size_t)D_ * H_ * DH_ * 2);
  half_t*        wv_h   = (half_t*)alloc((size_t)D_ * H_ * DH_ * 2);
  half_t*        wout_h = (half_t*)alloc((size_t)H_ * DH_ * D_ * 2);
  half_t*        wm1_h  = (half_t*)alloc((size_t)D_ * MLP_ * 2);
  half_t*        wm2_h  = (half_t*)alloc((size_t)MLP_ * D_ * 2);
  half_t*        qk_h   = (half_t*)alloc((size_t)BS_ * H_ * DH_ * 2);
  half_t*        v_h    = (half_t*)alloc((size_t)BS_ * H_ * DH_ * 2);
  half_t*        xn_h   = (half_t*)alloc((size_t)BS_ * H_ * DH_ * 2);
  unsigned short* bkt   = (unsigned short*)alloc((size_t)NROW_ * S_ * 2);
  unsigned int*  ordr   = (unsigned int*)alloc((size_t)NROW_ * S_ * 4);
  half_t*        o_h    = (half_t*)alloc((size_t)NROW_ * S_ * DH_ * 2);
  float*         slog   = (float*)alloc((size_t)NROW_ * S_ * 4);
  half_t*        att_h  = (half_t*)alloc((size_t)BS_ * H_ * DH_ * 2);
  float*         x2     = (float*)alloc((size_t)BS_ * D_ * 4);
  half_t*        y2     = (half_t*)alloc((size_t)BS_ * D_ * 2);
  half_t*        h1     = (half_t*)alloc((size_t)BS_ * MLP_ * 2);

  // weights -> f16 (B operands, already [K,N] row-major as stored)
  int n;
  n = D_ * H_ * DH_;
  f32_to_f16_kernel<<<(n + 255) / 256, 256, 0, stream>>>(w_qk, wqk_h, n);
  f32_to_f16_kernel<<<(n + 255) / 256, 256, 0, stream>>>(w_v, wv_h, n);
  n = H_ * DH_ * D_;
  f32_to_f16_kernel<<<(n + 255) / 256, 256, 0, stream>>>(w_out, wout_h, n);
  n = D_ * MLP_;
  f32_to_f16_kernel<<<(n + 255) / 256, 256, 0, stream>>>(w_mlp1, wm1_h, n);
  n = MLP_ * D_;
  f32_to_f16_kernel<<<(n + 255) / 256, 256, 0, stream>>>(w_mlp2, wm2_h, n);

  // LN1
  ln_kernel<<<BS_, 256, 0, stream>>>(inputs, ln1_scale, ln1_bias, xln);

  // QK / V projections: [8192,1024] x [1024,1024]
  dim3 gProj(D_ / 128, BS_ / 128);
  gemm_wmma_kernel<<<gProj, 256, 0, stream>>>(xln, wqk_h, nullptr, nullptr,
                                              nullptr, qk_h, BS_, H_ * DH_, D_,
                                              MODE_F16);
  gemm_wmma_kernel<<<gProj, 256, 0, stream>>>(xln, wv_h, nullptr, nullptr,
                                              nullptr, v_h, BS_, H_ * DH_, D_,
                                              MODE_F16);

  // LSH bucketing + stable per-row sort
  int nth = B_ * H_ * S_;
  bucket_kernel<<<nth / 256, 256, 0, stream>>>(qk_h, rotations, xn_h, bkt);
  sort_kernel<<<NROW_, 256, 0, stream>>>(bkt, ordr);

  // chunked attention (one block per row x chunk)
  dim3 gAtt(NC_, NROW_);
  attn_kernel<<<gAtt, 256, 0, stream>>>(qk_h, xn_h, v_h, ordr, o_h, slog);

  // combine hash rounds by logsumexp softmax
  combine_kernel<<<nth / 256, 256, 0, stream>>>(o_h, slog, att_h);

  // out projection + residual: x2 = inputs + att @ w_out
  gemm_wmma_kernel<<<gProj, 256, 0, stream>>>(att_h, wout_h, nullptr, inputs, x2,
                                              nullptr, BS_, D_, H_ * DH_,
                                              MODE_RES_F32);

  // LN2 + MLP
  ln_kernel<<<BS_, 256, 0, stream>>>(x2, ln2_scale, ln2_bias, y2);
  dim3 gM1(MLP_ / 128, BS_ / 128);
  gemm_wmma_kernel<<<gM1, 256, 0, stream>>>(y2, wm1_h, b_mlp1, nullptr, nullptr,
                                            h1, BS_, MLP_, D_, MODE_BIAS_GELU_F16);
  dim3 gM2(D_ / 128, BS_ / 128);
  gemm_wmma_kernel<<<gM2, 256, 0, stream>>>(h1, wm2_h, b_mlp2, x2, out, nullptr,
                                            BS_, D_, MLP_, MODE_BIAS_RES_F32);
}
